// QMatMul_52140902974013
// MI455X (gfx1250) — matrix operand order repaired and checked by reading, not compiled, not run on hardware
//
#include <hip/hip_runtime.h>

// ---------------------------------------------------------------------------
// QMatMul: out = A @ B  (scales cancel: (A/sA)@(B/sB)*(sA*sB) == A@B)
// A: [4,16,2048,64] f32   B: [4,16,64,2048] f32
// out: [4,16,2048,2048] f32  ++  [1] f32 (sA*sB)
//
// bf16x3 split-product WMMA (hi*hi + hi*lo + lo*hi) in f32 accumulators via
// v_wmma_f32_16x16x32_bf16. If d_ws is big enough (64 MB), a pre-pass
// converts A -> (Ah,Al) and B -> transposed (Bth,Btl) bf16 planes ONCE, so the
// GEMM staging is a pure tile copy — done with GLOBAL_LOAD_ASYNC_TO_LDS_B128
// (ASYNCcnt-tracked, no VGPR round trip) when the builtin is available.
// Epilogue restages the 128x128 f32 tile through LDS for b128 stores.
// ---------------------------------------------------------------------------

typedef __attribute__((ext_vector_type(16))) __bf16 v16bf;
typedef __attribute__((ext_vector_type(8)))  __bf16 v8bf;
typedef __attribute__((ext_vector_type(8)))  float  v8f;

#define BH_COUNT 64
#define M_DIM 2048
#define N_DIM 2048
#define K_DIM 64
#define TILE_M 128
#define TILE_N 128

#if defined(__has_builtin)
#if __has_builtin(__builtin_amdgcn_global_load_async_to_lds_b128)
#define HAVE_ASYNC_LDS 1
#endif
#endif

#ifdef HAVE_ASYNC_LDS
// Exact parameter element type per the builtin's signature: int __vector_size__(16).
typedef __attribute__((__vector_size__(4 * sizeof(int)))) int v4i;
typedef __attribute__((address_space(1))) v4i as1_v4i;   // global
typedef __attribute__((address_space(3))) v4i as3_v4i;   // LDS
#endif

__device__ __forceinline__ void wait_async_zero() {
#if defined(__has_builtin) && __has_builtin(__builtin_amdgcn_s_wait_asynccnt)
    __builtin_amdgcn_s_wait_asynccnt(0);
#else
    asm volatile("s_wait_asynccnt 0x0" ::: "memory");
#endif
}

__device__ __forceinline__ unsigned short f32_to_bf16_rne(float f) {
    unsigned int u = __float_as_uint(f);
    u += 0x7FFFu + ((u >> 16) & 1u);   // round-to-nearest-even (finite inputs)
    return (unsigned short)(u >> 16);
}
__device__ __forceinline__ float bf16_bits_to_f32(unsigned short h) {
    return __uint_as_float(((unsigned int)h) << 16);
}

// ---------------- Pre-pass: A -> Ah/Al planes (linear) ----------------------
__global__ __launch_bounds__(256)
void convert_A_kernel(const float* __restrict__ A,
                      unsigned short* __restrict__ Ah,
                      unsigned short* __restrict__ Al) {
    const size_t i = (size_t)blockIdx.x * 256 + threadIdx.x;  // float4 index
    const float4 v = ((const float4*)A)[i];
    ushort4 hv, lv;
    hv.x = f32_to_bf16_rne(v.x); lv.x = f32_to_bf16_rne(v.x - bf16_bits_to_f32(hv.x));
    hv.y = f32_to_bf16_rne(v.y); lv.y = f32_to_bf16_rne(v.y - bf16_bits_to_f32(hv.y));
    hv.z = f32_to_bf16_rne(v.z); lv.z = f32_to_bf16_rne(v.z - bf16_bits_to_f32(hv.z));
    hv.w = f32_to_bf16_rne(v.w); lv.w = f32_to_bf16_rne(v.w - bf16_bits_to_f32(hv.w));
    *(ushort4*)&Ah[i * 4] = hv;
    *(ushort4*)&Al[i * 4] = lv;
}

// ------- Pre-pass: B[k][n] -> Bt[n][k] hi/lo planes via 64x64 LDS tile ------
__global__ __launch_bounds__(256)
void convert_B_kernel(const float* __restrict__ B,
                      unsigned short* __restrict__ Bth,
                      unsigned short* __restrict__ Btl) {
    __shared__ __attribute__((aligned(16))) unsigned short th[64 * 64];
    __shared__ __attribute__((aligned(16))) unsigned short tl[64 * 64];
    const int tid = threadIdx.x;
    const int bh  = blockIdx.y;
    const int n0  = blockIdx.x * 64;

    const float* Bb = B + (size_t)bh * (K_DIM * N_DIM) + n0;
    #pragma unroll
    for (int i = tid; i < (64 * 64) / 4; i += 256) {   // coalesced float4 reads
        const int k  = i >> 4;                          // 16 float4 per 64-wide row
        const int c4 = i & 15;
        const float4 v = *(const float4*)(Bb + (size_t)k * N_DIM + c4 * 4);
        const int n = c4 * 4;
        unsigned short h0 = f32_to_bf16_rne(v.x);
        unsigned short h1 = f32_to_bf16_rne(v.y);
        unsigned short h2 = f32_to_bf16_rne(v.z);
        unsigned short h3 = f32_to_bf16_rne(v.w);
        th[(n + 0) * 64 + k] = h0;
        th[(n + 1) * 64 + k] = h1;
        th[(n + 2) * 64 + k] = h2;
        th[(n + 3) * 64 + k] = h3;
        tl[(n + 0) * 64 + k] = f32_to_bf16_rne(v.x - bf16_bits_to_f32(h0));
        tl[(n + 1) * 64 + k] = f32_to_bf16_rne(v.y - bf16_bits_to_f32(h1));
        tl[(n + 2) * 64 + k] = f32_to_bf16_rne(v.z - bf16_bits_to_f32(h2));
        tl[(n + 3) * 64 + k] = f32_to_bf16_rne(v.w - bf16_bits_to_f32(h3));
    }
    __syncthreads();
    // Destination [bh][n0..n0+63][0..63] is one contiguous 8 KB block per plane.
    unsigned short* dh = Bth + ((size_t)bh * N_DIM + n0) * K_DIM;
    unsigned short* dl = Btl + ((size_t)bh * N_DIM + n0) * K_DIM;
    #pragma unroll
    for (int i = tid; i < (64 * 64) / 4; i += 256) {   // coalesced ushort4 writes
        *(ushort4*)&dh[i * 4] = ((const ushort4*)th)[i];
        *(ushort4*)&dl[i * 4] = ((const ushort4*)tl)[i];
    }
}

// ----------------------------- Main GEMM ------------------------------------
template <bool PRECONV>
__global__ __launch_bounds__(256)
void qmatmul_kernel(const float* __restrict__ A, const float* __restrict__ B,
                    const unsigned short* __restrict__ Ah,
                    const unsigned short* __restrict__ Al,
                    const unsigned short* __restrict__ Bth,
                    const unsigned short* __restrict__ Btl,
                    float* __restrict__ C) {
    // 64 KB LDS: bf16 staging planes, later reused as the f32 output tile.
    __shared__ __attribute__((aligned(16))) unsigned char smem[65536];
    unsigned short* sAhi = (unsigned short*)(smem);
    unsigned short* sAlo = (unsigned short*)(smem + 16384);
    unsigned short* sBhi = (unsigned short*)(smem + 32768);
    unsigned short* sBlo = (unsigned short*)(smem + 49152);
    float*          sOut = (float*)(smem);

    const int tid  = threadIdx.x;
    const int bhid = blockIdx.z;
    const int m0b  = blockIdx.y * TILE_M;
    const int n0b  = blockIdx.x * TILE_N;

    if (PRECONV) {
        // Pure tile copy: each plane tile is one contiguous 16 KB block.
        const unsigned short* src[4] = {
            Ah  + ((size_t)bhid * M_DIM + m0b) * K_DIM,
            Al  + ((size_t)bhid * M_DIM + m0b) * K_DIM,
            Bth + ((size_t)bhid * N_DIM + n0b) * K_DIM,
            Btl + ((size_t)bhid * N_DIM + n0b) * K_DIM };
        unsigned short* dst[4] = { sAhi, sAlo, sBhi, sBlo };
        #pragma unroll
        for (int p = 0; p < 4; ++p) {
            const uint4* s4 = (const uint4*)src[p];
            uint4*       d4 = (uint4*)dst[p];
            #pragma unroll
            for (int i = tid; i < (TILE_M * K_DIM) / 8; i += 256) {  // 1024 x 16B
#ifdef HAVE_ASYNC_LDS
                // 16 B/lane directly global -> LDS, ASYNCcnt-tracked.
                __builtin_amdgcn_global_load_async_to_lds_b128(
                    (as1_v4i*)(s4 + i), (as3_v4i*)(d4 + i), 0, 0);
#else
                d4[i] = s4[i];
#endif
            }
        }
#ifdef HAVE_ASYNC_LDS
        wait_async_zero();   // this wave's tiles landed; barrier covers the rest
#endif
    } else {
        // Fused conversion fallback (used when d_ws is too small).
        const float4* A4 = (const float4*)(A + (size_t)bhid * (M_DIM * K_DIM)
                                             + (size_t)m0b * K_DIM);
        #pragma unroll
        for (int i = tid; i < (TILE_M * K_DIM) / 4; i += 256) {
            const float4 v = A4[i];
            ushort4 hv, lv;
            hv.x = f32_to_bf16_rne(v.x); lv.x = f32_to_bf16_rne(v.x - bf16_bits_to_f32(hv.x));
            hv.y = f32_to_bf16_rne(v.y); lv.y = f32_to_bf16_rne(v.y - bf16_bits_to_f32(hv.y));
            hv.z = f32_to_bf16_rne(v.z); lv.z = f32_to_bf16_rne(v.z - bf16_bits_to_f32(hv.z));
            hv.w = f32_to_bf16_rne(v.w); lv.w = f32_to_bf16_rne(v.w - bf16_bits_to_f32(hv.w));
            *(ushort4*)&sAhi[i * 4] = hv;
            *(ushort4*)&sAlo[i * 4] = lv;
        }
        const float* Bb = B + (size_t)bhid * (K_DIM * N_DIM) + n0b;
        #pragma unroll
        for (int i = tid; i < (K_DIM * TILE_N) / 4; i += 256) {
            const int k  = i >> 5;
            const int c4 = i & 31;
            const float4 v = *(const float4*)(Bb + (size_t)k * N_DIM + c4 * 4);
            const int n = c4 * 4;
            unsigned short h0 = f32_to_bf16_rne(v.x);
            unsigned short h1 = f32_to_bf16_rne(v.y);
            unsigned short h2 = f32_to_bf16_rne(v.z);
            unsigned short h3 = f32_to_bf16_rne(v.w);
            sBhi[(n + 0) * K_DIM + k] = h0;
            sBhi[(n + 1) * K_DIM + k] = h1;
            sBhi[(n + 2) * K_DIM + k] = h2;
            sBhi[(n + 3) * K_DIM + k] = h3;
            sBlo[(n + 0) * K_DIM + k] = f32_to_bf16_rne(v.x - bf16_bits_to_f32(h0));
            sBlo[(n + 1) * K_DIM + k] = f32_to_bf16_rne(v.y - bf16_bits_to_f32(h1));
            sBlo[(n + 2) * K_DIM + k] = f32_to_bf16_rne(v.z - bf16_bits_to_f32(h2));
            sBlo[(n + 3) * K_DIM + k] = f32_to_bf16_rne(v.w - bf16_bits_to_f32(h3));
        }
    }

    __syncthreads();

    // ---- Compute: 8 waves in 4(M) x 2(N); each wave owns 32x64 = 2x4 tiles.
    const int lane = tid & 31;
    const int w    = tid >> 5;
    const int m0   = (w >> 1) * 32;
    const int n0   = (w & 1) * 64;
    const int r    = lane & 15;
    const int h    = lane >> 4;

    v8f acc[2][4];
    #pragma unroll
    for (int mt = 0; mt < 2; ++mt)
        #pragma unroll
        for (int nt = 0; nt < 4; ++nt)
            acc[mt][nt] = (v8f){0.f, 0.f, 0.f, 0.f, 0.f, 0.f, 0.f, 0.f};

    #pragma unroll
    for (int kc = 0; kc < 2; ++kc) {
        // A fragments: lane r = row m; K = [16h, 16h+16) contiguous.
        v16bf afh[2], afl[2];
        #pragma unroll
        for (int mt = 0; mt < 2; ++mt) {
            const int base = (m0 + mt * 16 + r) * K_DIM + kc * 32 + 16 * h;
            const v8bf h0 = *(const v8bf*)&sAhi[base];
            const v8bf h1 = *(const v8bf*)&sAhi[base + 8];
            afh[mt] = __builtin_shufflevector(h0, h1, 0,1,2,3,4,5,6,7,8,9,10,11,12,13,14,15);
            const v8bf l0 = *(const v8bf*)&sAlo[base];
            const v8bf l1 = *(const v8bf*)&sAlo[base + 8];
            afl[mt] = __builtin_shufflevector(l0, l1, 0,1,2,3,4,5,6,7,8,9,10,11,12,13,14,15);
        }
        // B fragments: lane r = column n; K = [16h, 16h+16) contiguous.
        v16bf bfh[4], bfl[4];
        #pragma unroll
        for (int nt = 0; nt < 4; ++nt) {
            const int base = (n0 + nt * 16 + r) * K_DIM + kc * 32 + 16 * h;
            bfh[nt] = *(const v16bf*)&sBhi[base];
            bfl[nt] = *(const v16bf*)&sBlo[base];
        }
        // bf16x3: acc += ah*bh + ah*bl + al*bh
        #pragma unroll
        for (int mt = 0; mt < 2; ++mt) {
            #pragma unroll
            for (int nt = 0; nt < 4; ++nt) {
                acc[mt][nt] = __builtin_amdgcn_wmma_f32_16x16x32_bf16(
                    false, afh[mt], false, bfh[nt], (short)0, acc[mt][nt], false, false);
                acc[mt][nt] = __builtin_amdgcn_wmma_f32_16x16x32_bf16(
                    false, afh[mt], false, bfl[nt], (short)0, acc[mt][nt], false, false);
                acc[mt][nt] = __builtin_amdgcn_wmma_f32_16x16x32_bf16(
                    false, afl[mt], false, bfh[nt], (short)0, acc[mt][nt], false, false);
            }
        }
    }

    // ---- Epilogue: restage tile through LDS, then coalesced b128 stores.
    __syncthreads();   // everyone done reading bf16 planes; reuse LDS as f32 tile
    #pragma unroll
    for (int mt = 0; mt < 2; ++mt) {
        const int rowBase = m0 + mt * 16 + 8 * h;       // tile-local row
        #pragma unroll
        for (int nt = 0; nt < 4; ++nt) {
            const int col = n0 + nt * 16 + r;           // tile-local col
            #pragma unroll
            for (int v = 0; v < 8; ++v)
                sOut[(rowBase + v) * TILE_N + col] = acc[mt][nt][v];
        }
    }
    __syncthreads();

    float* Cb = C + (size_t)bhid * ((size_t)M_DIM * N_DIM);
    #pragma unroll
    for (int i = tid; i < (TILE_M * TILE_N) / 4; i += 256) {  // 512B/wave bursts
        const int row  = i >> 5;                               // 32 float4 per row
        const int col4 = i & 31;
        *(float4*)(Cb + (size_t)(m0b + row) * N_DIM + n0b + col4 * 4) =
            *(const float4*)&sOut[row * TILE_N + col4 * 4];
    }
}

__global__ void qmatmul_scale_kernel(const float* __restrict__ sA,
                                     const float* __restrict__ sB,
                                     float* __restrict__ out_scale) {
    if (threadIdx.x == 0 && blockIdx.x == 0) {
        out_scale[0] = sA[0] * sB[0];
    }
}

extern "C" void kernel_launch(void* const* d_in, const int* in_sizes, int n_in,
                              void* d_out, int out_size, void* d_ws, size_t ws_size,
                              hipStream_t stream) {
    const float* A  = (const float*)d_in[0];
    const float* sA = (const float*)d_in[1];
    const float* B  = (const float*)d_in[2];
    const float* sB = (const float*)d_in[3];
    float* out = (float*)d_out;

    const size_t PLANE = (size_t)BH_COUNT * M_DIM * K_DIM;   // elements per plane
    const size_t WS_NEEDED = 4 * PLANE * sizeof(unsigned short);  // 64 MB
    const bool use_ws = (d_ws != nullptr) && (ws_size >= WS_NEEDED);

    dim3 grid(N_DIM / TILE_N, M_DIM / TILE_M, BH_COUNT);  // (16,16,64)
    dim3 block(256);

    if (use_ws) {
        unsigned short* Ah  = (unsigned short*)d_ws;
        unsigned short* Al  = Ah + PLANE;
        unsigned short* Bth = Ah + 2 * PLANE;
        unsigned short* Btl = Ah + 3 * PLANE;
        convert_A_kernel<<<dim3((unsigned)(PLANE / 1024)), block, 0, stream>>>(A, Ah, Al);
        convert_B_kernel<<<dim3(N_DIM / 64, BH_COUNT), block, 0, stream>>>(B, Bth, Btl);
        qmatmul_kernel<true><<<grid, block, 0, stream>>>(A, B, Ah, Al, Bth, Btl, out);
    } else {
        qmatmul_kernel<false><<<grid, block, 0, stream>>>(
            A, B, nullptr, nullptr, nullptr, nullptr, out);
    }

    float* out_scale = out + (size_t)BH_COUNT * M_DIM * N_DIM;
    qmatmul_scale_kernel<<<dim3(1), dim3(32), 0, stream>>>(sA, sB, out_scale);
}
